// SwinTransformerBlock_76312978916030
// MI455X (gfx1250) — matrix operand order repaired and checked by reading, not compiled, not run on hardware
//
#include <hip/hip_runtime.h>
#include <hip/hip_bf16.h>
#include <math.h>

// ---------------------------------------------------------------------------
// Types for CDNA5 WMMA (wave32): v_wmma_f32_16x16x32_bf16
// ---------------------------------------------------------------------------
typedef __bf16 bf16_t;
typedef __attribute__((ext_vector_type(16))) __bf16 v16bf;
typedef __attribute__((ext_vector_type(8)))  __bf16 v8bf;
typedef __attribute__((ext_vector_type(8)))  float  v8f;

static constexpr int CB        = 192;          // channels
static constexpr int HB        = 768;          // MLP hidden
static constexpr int HEADS_    = 6;
static constexpr int HD        = 32;           // head dim
static constexpr int IMG_H     = 128;
static constexpr int IMG_W     = 256;
static constexpr int BATCH     = 4;
static constexpr int TOK       = BATCH * IMG_H * IMG_W;   // 131072
static constexpr int NWIN      = TOK / 64;                // 2048 windows
static constexpr int WIN_PER_IMG = (IMG_H / 8) * (IMG_W / 8); // 512
static constexpr int WIN_PER_ROW = IMG_W / 8;             // 32
static constexpr int SHIFT     = 4;

// ---------------------------------------------------------------------------
// Fragment loaders (wave32): contiguous k = 16g + s for both A and B, not the ISA table order.
// A (16x32 bf16, row-major MxK):  lane L: g=L/16, m=L%16
//   halves 0..7  -> K = k0 + 16g .. +7       (16 contiguous bytes)
//   halves 8..15 -> K = k0 + 16g + 8 .. +7   (16 contiguous bytes)
// B (32x16 bf16) loaded from Bt stored [N][K]: lane L: g=L/16, n=L%16
//   halves 0..15 -> K = k0 + 16g .. +15      (32 contiguous bytes)
// ---------------------------------------------------------------------------
__device__ __forceinline__ v16bf frag_a(const bf16_t* A, int lda, int row0, int k0) {
  int lane = threadIdx.x & 31;
  int g = lane >> 4, m = lane & 15;
  const bf16_t* p = A + (size_t)(row0 + m) * lda + k0 + 16 * g;
  v8bf lo = *(const v8bf*)p;
  v8bf hi = *(const v8bf*)(p + 8);
  v16bf r;
#pragma unroll
  for (int i = 0; i < 8; ++i) { r[i] = lo[i]; r[i + 8] = hi[i]; }
  return r;
}

__device__ __forceinline__ v16bf frag_b(const bf16_t* Bt, int ldb, int n0, int k0) {
  int lane = threadIdx.x & 31;
  int g = lane >> 4, n = lane & 15;
  const bf16_t* p = Bt + (size_t)(n0 + n) * ldb + k0 + 16 * g;
  return *(const v16bf*)p;
}

__device__ __forceinline__ v8f wmma_bf16(v16bf a, v16bf b, v8f c) {
  // 8 args: (neg_a, A, neg_b, B, c_mod, C, reuse_a, reuse_b)
  return __builtin_amdgcn_wmma_f32_16x16x32_bf16(false, a, false, b, (short)0, c,
                                                 false, false);
}

// ---------------------------------------------------------------------------
// Async stage of a 64(N) x 64(K) bf16 panel chunk into LDS via the CDNA5
// async-DMA path (tracked by ASYNCcnt). 128 threads x 4 iters = 512 x b128.
// ---------------------------------------------------------------------------
__device__ __forceinline__ void stage_panel(const bf16_t* Bt, int K, int n_base,
                                            int kc, bf16_t* dstbase, int tid) {
  uint32_t base = (uint32_t)(uintptr_t)dstbase;   // LDS byte offset (addr[31:0])
#pragma unroll
  for (int r = 0; r < 4; ++r) {
    int id = tid + r * 128;                       // 0..511
    int row = id >> 3, seg = id & 7;              // row 0..63, seg 0..7 (8 elems)
    const bf16_t* gp = Bt + (size_t)(n_base + row) * K + kc * 64 + seg * 8;
    uint32_t loff = base + (uint32_t)(row * 64 + seg * 8) * 2;
    asm volatile("global_load_async_to_lds_b128 %0, %1, off"
                 :: "v"(loff), "v"((unsigned long long)(uintptr_t)gp)
                 : "memory");
  }
}

// ---------------------------------------------------------------------------
// Weight convert + transpose: dst[N][K] (bf16) = src[K][N] (f32)
// ---------------------------------------------------------------------------
__global__ void wtrans_kernel(const float* __restrict__ src, bf16_t* __restrict__ dst,
                              int K, int N) {
  int i = blockIdx.x * 256 + threadIdx.x;
  if (i >= K * N) return;
  int k = i / N, n = i - k * N;
  dst[(size_t)n * K + k] = (bf16_t)src[i];
}

// ---------------------------------------------------------------------------
// LayerNorm (wave per token, 192 = 6 per lane). shifted==1 fuses the
// cyclic-shift + window-partition gather for the attention input.
// ---------------------------------------------------------------------------
__global__ __launch_bounds__(256) void ln_kernel(const float* __restrict__ in,
                                                 const float* __restrict__ gamma,
                                                 const float* __restrict__ beta,
                                                 bf16_t* __restrict__ out, int shifted) {
  int wave = threadIdx.x >> 5;
  int lane = threadIdx.x & 31;
  int row = blockIdx.x * 8 + wave;   // output row index
  int src;
  if (shifted) {
    int win = row >> 6, t = row & 63;
    int img = win / WIN_PER_IMG, wl = win - img * WIN_PER_IMG;
    int wy = wl / WIN_PER_ROW, wx = wl - wy * WIN_PER_ROW;
    int ty = t >> 3, tx = t & 7;
    int h = (wy * 8 + ty + SHIFT) & (IMG_H - 1);
    int w = (wx * 8 + tx + SHIFT) & (IMG_W - 1);
    src = img * (IMG_H * IMG_W) + h * IMG_W + w;
  } else {
    src = row;
  }
  const float* p = in + (size_t)src * CB;
  float v[6];
  float s = 0.f, s2 = 0.f;
#pragma unroll
  for (int i = 0; i < 6; ++i) {
    v[i] = p[lane + 32 * i];
    s += v[i];
    s2 += v[i] * v[i];
  }
#pragma unroll
  for (int m = 16; m >= 1; m >>= 1) {
    s  += __shfl_xor(s,  m, 32);
    s2 += __shfl_xor(s2, m, 32);
  }
  float mu  = s * (1.f / CB);
  float var = s2 * (1.f / CB) - mu * mu;
  float inv = rsqrtf(var + 1e-5f);
  bf16_t* q = out + (size_t)row * CB;
#pragma unroll
  for (int i = 0; i < 6; ++i) {
    int c = lane + 32 * i;
    q[c] = (bf16_t)((v[i] - mu) * inv * gamma[c] + beta[c]);
  }
}

// ---------------------------------------------------------------------------
// Shared-weight bf16 WMMA GEMM with async double-buffered B panels in LDS.
// C[M][N] = A[M][K] @ Bt[N][K]^T + bias. 4 waves/block, wave owns 16x64 strip.
// mode 0: q    -> (val+b)*scale scattered to q[win][head][64][32]  (bf16)
// mode 1: kv   -> k[win][head][64][32], vT[win][head][32][64]      (bf16)
// mode 2: proj -> window-reverse + roll + residual(x) into outf    (f32)
// ---------------------------------------------------------------------------
__global__ __launch_bounds__(128) void gemm_kernel(
    const bf16_t* __restrict__ A, const bf16_t* __restrict__ Bt,
    const float* __restrict__ bias, bf16_t* out_a, bf16_t* out_b, float* outf,
    const float* __restrict__ resid, int N, int K, int mode, float alpha) {
  __shared__ __align__(32) bf16_t bpan[2][64 * 64];
  int tid = threadIdx.x;
  int wave = tid >> 5;
  int lane = tid & 31;
  int g = lane >> 4, ln15 = lane & 15;
  int row0 = (blockIdx.y << 6) + (wave << 4);
  int n_base = blockIdx.x << 6;

  v8f acc[4];
#pragma unroll
  for (int j = 0; j < 4; ++j) { v8f z = {}; acc[j] = z; }

  int nk = K >> 6;                       // number of 64-wide K chunks
  stage_panel(Bt, K, n_base, 0, &bpan[0][0], tid);
  for (int kc = 0; kc < nk; ++kc) {
    if (kc + 1 < nk) {
      stage_panel(Bt, K, n_base, kc + 1, &bpan[(kc + 1) & 1][0], tid);
      asm volatile("s_wait_asynccnt 0x4" ::: "memory");  // current chunk done
    } else {
      asm volatile("s_wait_asynccnt 0x0" ::: "memory");
    }
    __syncthreads();
    const bf16_t* bp = &bpan[kc & 1][0];
#pragma unroll
    for (int kk = 0; kk < 64; kk += 32) {
      v16bf a = frag_a(A, K, row0, kc * 64 + kk);
#pragma unroll
      for (int j = 0; j < 4; ++j) {
        v16bf b = frag_b(bp, 64, 16 * j, kk);
        acc[j] = wmma_bf16(a, b, acc[j]);
      }
    }
    __syncthreads();                     // protect buffer about to be restaged
  }

#pragma unroll
  for (int j = 0; j < 4; ++j) {
    int col = n_base + 16 * j + ln15;
    float bv = bias[col];
#pragma unroll
    for (int e = 0; e < 8; ++e) {
      int row = row0 + e + 8 * g;        // C/D layout: M = e + 8g
      float val = acc[j][e] + bv;
      if (mode == 0) {                   // ---- q (scale folded in)
        val *= alpha;
        int win = row >> 6, t = row & 63;
        int head = col >> 5, d = col & 31;
        out_a[(((size_t)win * HEADS_ + head) * 64 + t) * HD + d] = (bf16_t)val;
      } else if (mode == 1) {            // ---- kv
        int win = row >> 6, t = row & 63;
        if (col < CB) {
          int head = col >> 5, d = col & 31;
          out_a[(((size_t)win * HEADS_ + head) * 64 + t) * HD + d] = (bf16_t)val;
        } else {
          int cc = col - CB;
          int head = cc >> 5, d = cc & 31;   // v stored transposed [head][32][64]
          out_b[(((size_t)win * HEADS_ + head) * HD + d) * 64 + t] = (bf16_t)val;
        }
      } else {                           // ---- proj + reverse + residual
        int win = row >> 6, t = row & 63;
        int img = win / WIN_PER_IMG, wl = win - img * WIN_PER_IMG;
        int wy = wl / WIN_PER_ROW, wx = wl - wy * WIN_PER_ROW;
        int ty = t >> 3, tx = t & 7;
        int h = (wy * 8 + ty + SHIFT) & (IMG_H - 1);
        int w = (wx * 8 + tx + SHIFT) & (IMG_W - 1);
        size_t idx = ((size_t)img * (IMG_H * IMG_W) + h * IMG_W + w) * CB + col;
        outf[idx] = resid[idx] + val;
      }
    }
  }
}

// ---------------------------------------------------------------------------
// Fused MLP: out += fc2(gelu(fc1(LN(xo)))). One wave per 16 rows; fc2
// accumulators (16x192) stay in VGPRs; each 64-wide hidden chunk is computed
// with WMMA, GELU'd, bounced through a per-wave LDS tile into A-fragment
// layout, and immediately consumed by the fc2 WMMAs. Kills the 400 MB hidden-
// tensor HBM round trip.
// ---------------------------------------------------------------------------
__global__ __launch_bounds__(128) void mlp_kernel(
    const bf16_t* __restrict__ A,      // LN(xo) bf16 [TOK][192]
    const bf16_t* __restrict__ W1t,    // [768][192]
    const bf16_t* __restrict__ W2t,    // [192][768]
    const float* __restrict__ b1, const float* __restrict__ b2,
    float* __restrict__ out) {         // f32 [TOK][192], read (xo) + write
  __shared__ __align__(32) bf16_t hlds[4][16 * 64];
  int wave = threadIdx.x >> 5;
  int lane = threadIdx.x & 31;
  int g = lane >> 4, ln15 = lane & 15;
  int rows0 = blockIdx.x * 64 + wave * 16;

  v8f acc2[12];
#pragma unroll
  for (int n = 0; n < 12; ++n) { v8f z = {}; acc2[n] = z; }

  for (int hc = 0; hc < HB / 64; ++hc) {
    // ---- fc1 chunk: h[16 x 64] = A[16 x 192] @ W1t[64 x 192]^T -------------
    v8f hacc[4];
#pragma unroll
    for (int j = 0; j < 4; ++j) { v8f z = {}; hacc[j] = z; }
    for (int k0 = 0; k0 < CB; k0 += 32) {
      v16bf a = frag_a(A, CB, rows0, k0);
#pragma unroll
      for (int j = 0; j < 4; ++j) {
        v16bf b = frag_b(W1t, CB, hc * 64 + 16 * j, k0);
        hacc[j] = wmma_bf16(a, b, hacc[j]);
      }
    }
    // ---- bias + exact GELU -> LDS (A-operand staging) ----------------------
#pragma unroll
    for (int j = 0; j < 4; ++j) {
      float bv = b1[hc * 64 + 16 * j + ln15];
#pragma unroll
      for (int e = 0; e < 8; ++e) {
        float v = hacc[j][e] + bv;
        float ge = 0.5f * v * (1.f + erff(v * 0.70710678118654752f));
        hlds[wave][(e + 8 * g) * 64 + 16 * j + ln15] = (bf16_t)ge;
      }
    }
    asm volatile("s_wait_dscnt 0x0" ::: "memory");  // same-wave LDS RAW fence
    // ---- fc2 accumulate: acc2 += h_chunk @ W2t[:, chunk]^T -----------------
    const bf16_t* pl = &hlds[wave][0];
#pragma unroll
    for (int kk = 0; kk < 64; kk += 32) {
      v16bf a2 = frag_a(pl, 64, 0, kk);
#pragma unroll
      for (int n = 0; n < 12; ++n) {
        v16bf b2 = frag_b(W2t, HB, 16 * n, hc * 64 + kk);
        acc2[n] = wmma_bf16(a2, b2, acc2[n]);
      }
    }
    // per-wave LDS reuse next hc is safe: DS ops are in-order within a wave
  }

#pragma unroll
  for (int n = 0; n < 12; ++n) {
    int col = 16 * n + ln15;
    float bv = b2[col];
#pragma unroll
    for (int e = 0; e < 8; ++e) {
      size_t idx = (size_t)(rows0 + e + 8 * g) * CB + col;
      out[idx] = out[idx] + bv + acc2[n][e];
    }
  }
}

// ---------------------------------------------------------------------------
// Attention: one wave per (window, head). 64x64 scores via WMMA, rel-pos bias
// from table, shift mask recomputed from window coords, shuffle softmax,
// P bounced through LDS into A-fragment layout, then P@V via WMMA.
// ---------------------------------------------------------------------------
__global__ __launch_bounds__(128) void attn_kernel(
    const bf16_t* __restrict__ qg, const bf16_t* __restrict__ kg,
    const bf16_t* __restrict__ vTg, const float* __restrict__ rpb,
    bf16_t* __restrict__ out) {
  __shared__ __align__(32) bf16_t plds[4][64 * 64];
  int wave = threadIdx.x >> 5;
  int lane = threadIdx.x & 31;
  int g = lane >> 4, ln15 = lane & 15;
  int wh = blockIdx.x * 4 + wave;            // win*HEADS + head
  int win = wh / HEADS_, head = wh - win * HEADS_;
  const bf16_t* qp = qg  + (size_t)wh * 64 * HD;
  const bf16_t* kp = kg  + (size_t)wh * 64 * HD;   // [key tok][32] == Bt layout
  const bf16_t* vp = vTg + (size_t)wh * HD * 64;   // [dim][tok]  == Bt layout

  int wl = win % WIN_PER_IMG;
  int wy = wl / WIN_PER_ROW, wx = wl - wy * WIN_PER_ROW;

  // -------- S = (q*scale) @ k^T : 4x4 tiles, single K=32 step --------------
  v8f s[4][4];
#pragma unroll
  for (int si = 0; si < 4; ++si) {
    v16bf a = frag_a(qp, HD, si * 16, 0);
#pragma unroll
    for (int sj = 0; sj < 4; ++sj) {
      v16bf b = frag_b(kp, HD, sj * 16, 0);
      v8f z = {};
      s[si][sj] = wmma_bf16(a, b, z);
    }
  }

  // -------- + rel-pos bias + shift mask, then row softmax ------------------
#pragma unroll
  for (int si = 0; si < 4; ++si) {
#pragma unroll
    for (int e = 0; e < 8; ++e) {
      int n = si * 16 + e + 8 * g;           // query token
      int tyn = n >> 3, txn = n & 7;
      int h2 = wy * 8 + tyn, w2 = wx * 8 + txn;
      int rh = (h2 < IMG_H - 8) ? 0 : ((h2 < IMG_H - 4) ? 1 : 2);
      int rw = (w2 < IMG_W - 8) ? 0 : ((w2 < IMG_W - 4) ? 1 : 2);
      int idn = rh * 3 + rw;
#pragma unroll
      for (int sj = 0; sj < 4; ++sj) {
        int m = sj * 16 + ln15;              // key token
        int tym = m >> 3, txm = m & 7;
        int h2m = wy * 8 + tym, w2m = wx * 8 + txm;
        int rhm = (h2m < IMG_H - 8) ? 0 : ((h2m < IMG_H - 4) ? 1 : 2);
        int rwm = (w2m < IMG_W - 8) ? 0 : ((w2m < IMG_W - 4) ? 1 : 2);
        int idm = rhm * 3 + rwm;
        int rel = (tyn - tym + 7) * 15 + (txn - txm + 7);
        float bias = rpb[rel * HEADS_ + head] + ((idn == idm) ? 0.f : -100.f);
        s[si][sj][e] += bias;
      }
      // row max / sum across 4 tiles x 16 lanes (xor 1,2,4,8 stays in half)
      float mx = -1e30f;
#pragma unroll
      for (int sj = 0; sj < 4; ++sj) mx = fmaxf(mx, s[si][sj][e]);
#pragma unroll
      for (int msk = 1; msk <= 8; msk <<= 1) mx = fmaxf(mx, __shfl_xor(mx, msk, 32));
      float sum = 0.f;
#pragma unroll
      for (int sj = 0; sj < 4; ++sj) {
        float p = __expf(s[si][sj][e] - mx);
        s[si][sj][e] = p;
        sum += p;
      }
#pragma unroll
      for (int msk = 1; msk <= 8; msk <<= 1) sum += __shfl_xor(sum, msk, 32);
      float inv = 1.f / sum;
#pragma unroll
      for (int sj = 0; sj < 4; ++sj) {
        int m = sj * 16 + ln15;
        plds[wave][n * 64 + m] = (bf16_t)(s[si][sj][e] * inv);
      }
    }
  }

  asm volatile("s_wait_dscnt 0x0" ::: "memory");  // same-wave LDS RAW fence

  // -------- O = P @ V : 4x2 tiles, K = 64 (two steps) ----------------------
  const bf16_t* pl = &plds[wave][0];
  v8f o[4][2];
#pragma unroll
  for (int si = 0; si < 4; ++si)
#pragma unroll
    for (int nj = 0; nj < 2; ++nj) { v8f z = {}; o[si][nj] = z; }
#pragma unroll
  for (int kk0 = 0; kk0 < 64; kk0 += 32) {
#pragma unroll
    for (int si = 0; si < 4; ++si) {
      v16bf a = frag_a(pl, 64, si * 16, kk0);
#pragma unroll
      for (int nj = 0; nj < 2; ++nj) {
        v16bf b = frag_b(vp, 64, nj * 16, kk0);
        o[si][nj] = wmma_bf16(a, b, o[si][nj]);
      }
    }
  }

  // -------- write [win*64 + tok][head*32 + dim] bf16 -----------------------
#pragma unroll
  for (int si = 0; si < 4; ++si)
#pragma unroll
    for (int nj = 0; nj < 2; ++nj)
#pragma unroll
      for (int e = 0; e < 8; ++e) {
        int tok = si * 16 + e + 8 * g;
        int dim = nj * 16 + ln15;
        out[((size_t)win * 64 + tok) * CB + head * HD + dim] = (bf16_t)o[si][nj][e];
      }
}

// ---------------------------------------------------------------------------
// Host launch
// ---------------------------------------------------------------------------
extern "C" void kernel_launch(void* const* d_in, const int* in_sizes, int n_in,
                              void* d_out, int out_size, void* d_ws, size_t ws_size,
                              hipStream_t stream) {
  (void)in_sizes; (void)n_in; (void)out_size; (void)ws_size;
  const float* x   = (const float*)d_in[0];
  const float* x1  = (const float*)d_in[1];
  const float* n1g = (const float*)d_in[3];
  const float* n1b = (const float*)d_in[4];
  const float* qw  = (const float*)d_in[5];
  const float* qbv = (const float*)d_in[6];
  const float* kvw = (const float*)d_in[7];
  const float* kvb = (const float*)d_in[8];
  const float* rpb = (const float*)d_in[9];
  const float* pw  = (const float*)d_in[10];
  const float* pbv = (const float*)d_in[11];
  const float* n2g = (const float*)d_in[12];
  const float* n2b = (const float*)d_in[13];
  const float* f1w = (const float*)d_in[14];
  const float* f1b = (const float*)d_in[15];
  const float* f2w = (const float*)d_in[16];
  const float* f2b = (const float*)d_in[17];
  float* out = (float*)d_out;                // doubles as fp32 residual xo

  char* ws = (char*)d_ws;
  auto carve = [&](size_t bytes) -> char* {
    char* p = ws;
    ws += (bytes + 255) & ~(size_t)255;
    return p;
  };
  bf16_t* wq   = (bf16_t*)carve((size_t)CB * CB * 2);
  bf16_t* wkv  = (bf16_t*)carve((size_t)CB * 2 * CB * 2);
  bf16_t* wp   = (bf16_t*)carve((size_t)CB * CB * 2);
  bf16_t* wf1  = (bf16_t*)carve((size_t)CB * HB * 2);
  bf16_t* wf2  = (bf16_t*)carve((size_t)HB * CB * 2);
  bf16_t* xw   = (bf16_t*)carve((size_t)TOK * CB * 2);
  bf16_t* x1w  = (bf16_t*)carve((size_t)TOK * CB * 2);
  bf16_t* qbuf = (bf16_t*)carve((size_t)TOK * CB * 2);
  bf16_t* kbuf = (bf16_t*)carve((size_t)TOK * CB * 2);
  bf16_t* vbuf = (bf16_t*)carve((size_t)TOK * CB * 2);
  bf16_t* aout = x1w;  // x1w dead after q GEMM
  bf16_t* xon  = xw;   // xw dead after kv GEMM

  // weight convert + transpose to [N][K] bf16
  wtrans_kernel<<<(CB * CB + 255) / 256, 256, 0, stream>>>(qw, wq, CB, CB);
  wtrans_kernel<<<(CB * 2 * CB + 255) / 256, 256, 0, stream>>>(kvw, wkv, CB, 2 * CB);
  wtrans_kernel<<<(CB * CB + 255) / 256, 256, 0, stream>>>(pw, wp, CB, CB);
  wtrans_kernel<<<(CB * HB + 255) / 256, 256, 0, stream>>>(f1w, wf1, CB, HB);
  wtrans_kernel<<<(HB * CB + 255) / 256, 256, 0, stream>>>(f2w, wf2, HB, CB);

  // LN + shifted window partition
  ln_kernel<<<TOK / 8, 256, 0, stream>>>(x,  n1g, n1b, xw,  1);
  ln_kernel<<<TOK / 8, 256, 0, stream>>>(x1, n1g, n1b, x1w, 1);

  const float scale = 0.17677669529663687f;  // 32^-0.5

  // q = (LN(x1)w @ q_w + q_b) * scale
  gemm_kernel<<<dim3(CB / 64, TOK / 64), 128, 0, stream>>>(
      x1w, wq, qbv, qbuf, nullptr, nullptr, nullptr, CB, CB, 0, scale);
  // kv = LN(x)w @ kv_w + kv_b  (k head-split, v transposed)
  gemm_kernel<<<dim3(2 * CB / 64, TOK / 64), 128, 0, stream>>>(
      xw, wkv, kvb, kbuf, vbuf, nullptr, nullptr, 2 * CB, CB, 1, 1.f);

  // windowed attention (2048 windows x 6 heads, 4 waves/block)
  attn_kernel<<<(NWIN * HEADS_) / 4, 128, 0, stream>>>(qbuf, kbuf, vbuf, rpb, aout);

  // proj + window reverse + roll + residual -> out (== xo, f32)
  gemm_kernel<<<dim3(CB / 64, TOK / 64), 128, 0, stream>>>(
      aout, wp, pbv, nullptr, nullptr, out, x, CB, CB, 2, 1.f);

  // norm2
  ln_kernel<<<TOK / 8, 256, 0, stream>>>(out, n2g, n2b, xon, 0);

  // fused MLP: out += fc2(gelu(fc1(xon)))
  mlp_kernel<<<TOK / 64, 128, 0, stream>>>(xon, wf1, wf2, f1b, f2b, out);
}
